// RGCN_19292993093709
// MI455X (gfx1250) — matrix operand order repaired and checked by reading, not compiled, not run on hardware
//
#include <hip/hip_runtime.h>
#include <stdint.h>

// ---------------- problem constants ----------------
#define N_NODES 4096
#define R_REL   8
#define B_BASIS 30
#define H_DIM   64
#define C_DIM   16
#define E_EDGES 131072
#define NCOL1   576   // R*H + H   (W1 blocks | root1)
#define NCOL2   144   // R*C + C   (W2 blocks | root2)

typedef __attribute__((ext_vector_type(16))) __bf16 bf16x16;
typedef __attribute__((ext_vector_type(8)))  float  f32x8;

typedef unsigned short u16;

// -------- async global->LDS staging (CDNA5 ASYNCcnt path), with fallback ----
#ifndef __has_builtin
#define __has_builtin(x) 0
#endif

#if __has_builtin(__builtin_amdgcn_global_load_async_to_lds_b128)
#define HAVE_ASYNC 1
// builtin signature: (as(1) int4* src, as(3) int4* dst, imm offset, imm cpol)
typedef int v4i_vec __attribute__((vector_size(16)));
typedef __attribute__((address_space(1))) v4i_vec* g_v4i_p;
typedef __attribute__((address_space(3))) v4i_vec* l_v4i_p;
// 16 bytes per lane, LDS written directly by the async engine (no VGPR round trip)
#define STAGE16(lp, gp)                                                        \
    __builtin_amdgcn_global_load_async_to_lds_b128((g_v4i_p)(gp), (l_v4i_p)(lp), 0, 0)
#else
#define HAVE_ASYNC 0
#define STAGE16(lp, gp) (*(uint4*)(lp) = *(const uint4*)(gp))
#endif

#if HAVE_ASYNC
#if __has_builtin(__builtin_amdgcn_s_wait_asynccnt)
#define WAIT_ASYNC() __builtin_amdgcn_s_wait_asynccnt(0)
#else
#define WAIT_ASYNC() asm volatile("s_wait_asynccnt 0x0" ::: "memory")
#endif
#else
#define WAIT_ASYNC() ((void)0)
#endif

// round-to-nearest-even float -> bf16 bit pattern (no intrinsic dependency)
__device__ __forceinline__ u16 f2bf(float f) {
    unsigned u = __float_as_uint(f);
    unsigned r = (u + 0x7FFFu + ((u >> 16) & 1u)) >> 16;
    return (u16)r;
}

// ---------------- small prep kernels ----------------
__global__ void k_zero(float* __restrict__ p, int n) {
    int i = blockIdx.x * blockDim.x + threadIdx.x;
    if (i < n) p[i] = 0.0f;
}

__global__ void k_cvt_x(const float4* __restrict__ x, uint2* __restrict__ xb) {
    int i = blockIdx.x * blockDim.x + threadIdx.x;   // one float4 -> 4 bf16
    float4 v = x[i];
    uint2 o;
    o.x = (unsigned)f2bf(v.x) | ((unsigned)f2bf(v.y) << 16);
    o.y = (unsigned)f2bf(v.z) | ((unsigned)f2bf(v.w) << 16);
    xb[i] = o;
}

// BmatT[col][k], col-major weight matrix for layer-1 GEMM (so the GEMM's B tile
// staging is a straight row-major copy, no transpose in the hot loop).
__global__ void k_bmat1(const float* __restrict__ basis1, const float* __restrict__ att1,
                        const float* __restrict__ root1, u16* __restrict__ bmatT) {
    __shared__ float att_s[R_REL * B_BASIS];
    int t = threadIdx.x;
    if (t < R_REL * B_BASIS) att_s[t] = att1[t];
    __syncthreads();
    int gid = blockIdx.x * blockDim.x + t;    // 4096*64 threads
    int k = gid >> 6, o = gid & 63;
    float acc[R_REL];
#pragma unroll
    for (int r = 0; r < R_REL; ++r) acc[r] = 0.0f;
    for (int b = 0; b < B_BASIS; ++b) {
        float v = basis1[((size_t)b * N_NODES + k) * H_DIM + o];   // coalesced
#pragma unroll
        for (int r = 0; r < R_REL; ++r) acc[r] += att_s[r * B_BASIS + b] * v;
    }
#pragma unroll
    for (int r = 0; r < R_REL; ++r)
        bmatT[(size_t)(r * H_DIM + o) * N_NODES + k] = f2bf(acc[r]);
    bmatT[(size_t)(R_REL * H_DIM + o) * N_NODES + k] = f2bf(root1[k * H_DIM + o]);
}

// W2t[col][k]: col<128 -> sum_b att2[r,b]*basis2[b,k,c] (r=col/16,c=col%16); col>=128 -> root2[k,col-128]
__global__ void k_bmat2(const float* __restrict__ basis2, const float* __restrict__ att2,
                        const float* __restrict__ root2, u16* __restrict__ w2t) {
    int t = blockIdx.x * blockDim.x + threadIdx.x;
    if (t >= NCOL2 * H_DIM) return;
    int col = t >> 6, k = t & 63;
    float v;
    if (col < R_REL * C_DIM) {
        int r = col >> 4, c = col & 15;
        v = 0.0f;
        for (int b = 0; b < B_BASIS; ++b)
            v += att2[r * B_BASIS + b] * basis2[((size_t)b * H_DIM + k) * C_DIM + c];
    } else {
        v = root2[k * C_DIM + (col - R_REL * C_DIM)];
    }
    w2t[(size_t)col * H_DIM + k] = f2bf(v);
}

__global__ void k_deg(const int* __restrict__ dst, float* __restrict__ deg) {
    int e = blockIdx.x * blockDim.x + threadIdx.x;
    if (e < E_EDGES) atomicAdd(&deg[dst[e]], 1.0f);
}

// ---------------- GEMM 1: [4096x4096]bf16 x [4096x576]bf16 -> f32 ----------------
// Block tile 256(M) x 64(N), 8 waves as 4(M) x 2(N), wave tile 64x32 (8 WMMA).
// Double-buffered LDS, async global->LDS staging, one barrier per K-step of 32.
#define BM 256
#define BN 64
#define BK 32
#define LDA_S 40   // padded LDS strides (elements)
#define LDB_S 40

__global__ __launch_bounds__(256) void k_gemm1(const u16* __restrict__ A,    // xb   [4096][4096]
                                               const u16* __restrict__ Bt,   // BmatT[576][4096]
                                               float* __restrict__ Cout) {   // h_all[4096][576]
    __shared__ u16 As[2][BM * LDA_S];   // [row][k]  2 x 20 KB
    __shared__ u16 Bs[2][BN * LDB_S];   // [col][k]  2 x  5 KB
    const int tid  = threadIdx.x;
    const int lane = tid & 31, wid = tid >> 5;
    const int wr = wid & 3, wc = wid >> 2;      // 4 (M) x 2 (N) waves
    const int m0 = blockIdx.x * BM;
    const int n0 = blockIdx.y * BN;
    const int lrow = lane & 15, half = lane >> 4;

    f32x8 acc[4][2] = {};

    // staging indices (256 threads)
    const int sa_row = tid >> 2, sa_kc = (tid & 3) * 8;   // A: 4 chunks/thread (rows +0,64,128,192)
    const int sb_row = tid >> 2, sb_kc = (tid & 3) * 8;   // B: 1 chunk/thread

    // per-lane fragment offsets in LDS (elements), lane half h holds K = 16h..16h+15 (same order for A and B)
    const int abase  = (wr * 64 + lrow) * LDA_S + half * 16;
    const int bbase0 = (wc * 32 + lrow) * LDB_S + half * 16;
    const int bbase1 = (wc * 32 + 16 + lrow) * LDB_S + half * 16;

    // ---- prologue: stage k0 = 0 into buffer 0 ----
#pragma unroll
    for (int i = 0; i < 4; ++i) {
        int row = sa_row + i * 64;
        STAGE16(&As[0][row * LDA_S + sa_kc], A + (size_t)(m0 + row) * 4096 + sa_kc);
    }
    STAGE16(&Bs[0][sb_row * LDB_S + sb_kc], Bt + (size_t)(n0 + sb_row) * 4096 + sb_kc);
    WAIT_ASYNC();
    __syncthreads();

    int cur = 0;
    for (int k0 = 0; k0 < 4096; k0 += BK) {
        const int nxt = cur ^ 1;
        // ---- stage next K-slice while computing current ----
        if (k0 + BK < 4096) {
            const int kn = k0 + BK;
#pragma unroll
            for (int i = 0; i < 4; ++i) {
                int row = sa_row + i * 64;
                STAGE16(&As[nxt][row * LDA_S + sa_kc], A + (size_t)(m0 + row) * 4096 + kn + sa_kc);
            }
            STAGE16(&Bs[nxt][sb_row * LDB_S + sb_kc], Bt + (size_t)(n0 + sb_row) * 4096 + kn + sb_kc);
        }

        // ---- compute on current buffer: 12 ds_load_b128, 8 WMMA ----
        union AB { bf16x16 v; uint4 u[2]; };
        AB b0, b1;
        b0.u[0] = *(const uint4*)(&Bs[cur][bbase0]);      // K = half*16 + 0..7
        b0.u[1] = *(const uint4*)(&Bs[cur][bbase0 + 8]);  // K = half*16 + 8..15
        b1.u[0] = *(const uint4*)(&Bs[cur][bbase1]);
        b1.u[1] = *(const uint4*)(&Bs[cur][bbase1 + 8]);
#pragma unroll
        for (int i = 0; i < 4; ++i) {
            AB a;
            const int ao = abase + i * 16 * LDA_S;
            a.u[0] = *(const uint4*)(&As[cur][ao]);       // K = half*16 + 0..7
            a.u[1] = *(const uint4*)(&As[cur][ao + 8]);   // K = half*16 + 8..15
            acc[i][0] = __builtin_amdgcn_wmma_f32_16x16x32_bf16(false, a.v, false, b0.v,
                                                                (short)0, acc[i][0], false, false);
            acc[i][1] = __builtin_amdgcn_wmma_f32_16x16x32_bf16(false, a.v, false, b1.v,
                                                                (short)0, acc[i][1], false, false);
        }
        WAIT_ASYNC();          // staged buffer fully in LDS before all waves pass barrier
        __syncthreads();
        cur = nxt;
    }

    // C layout: VGPR v, lanes 0-15: M=v, N=lane; lanes 16-31: M=v+8, N=lane-16
#pragma unroll
    for (int i = 0; i < 4; ++i) {
        const int rb = m0 + wr * 64 + i * 16;
#pragma unroll
        for (int j = 0; j < 2; ++j) {
            const int cb = n0 + wc * 32 + j * 16;
#pragma unroll
            for (int v = 0; v < 8; ++v) {
                int row = rb + v + half * 8;
                Cout[(size_t)row * NCOL1 + cb + lrow] = acc[i][j][v];
            }
        }
    }
}

// ---------------- edge scatter (layer 1): wave per edge, 64 floats ----------------
__global__ void k_scatter1(const int* __restrict__ src, const int* __restrict__ dst,
                           const int* __restrict__ et, const float* __restrict__ h_all,
                           float* __restrict__ agg1) {
    int g = blockIdx.x * blockDim.x + threadIdx.x;
    int e = g >> 5, lane = g & 31;
    if (e >= E_EDGES) return;
    int s = src[e], d = dst[e], r = et[e];
    const float* hp = h_all + (size_t)s * NCOL1 + r * H_DIM;
    float* ap = agg1 + (size_t)d * H_DIM;
    atomicAdd(ap + lane,      hp[lane]);
    atomicAdd(ap + lane + 32, hp[lane + 32]);
}

__global__ void k_fin1(const float* __restrict__ agg1, const float* __restrict__ h_all,
                       const float* __restrict__ deg, const float* __restrict__ bias1,
                       u16* __restrict__ h1bf) {
    int g = blockIdx.x * blockDim.x + threadIdx.x;   // 4096*64
    int n = g >> 6, o = g & 63;
    float di = 1.0f / fmaxf(deg[n], 1.0f);
    float v = agg1[g] * di + h_all[(size_t)n * NCOL1 + R_REL * H_DIM + o] + bias1[o];
    h1bf[g] = f2bf(fmaxf(v, 0.0f));
}

// ---------------- GEMM 2: [4096x64]bf16 x [64x144]bf16 -> f32 (no LDS) ----------------
__global__ __launch_bounds__(256) void k_gemm2(const u16* __restrict__ A,   // h1 [4096][64]
                                               const u16* __restrict__ Bt,  // w2t [144][64]
                                               float* __restrict__ Cout) {  // [4096][144]
    const int tid  = threadIdx.x;
    const int lane = tid & 31, wid = tid >> 5;
    const int lrow = lane & 15, half = lane >> 4;
    const int rowbase = blockIdx.x * 128 + wid * 16;
    const int colbase = blockIdx.y * 16;
    f32x8 c = {};
#pragma unroll
    for (int k0 = 0; k0 < 64; k0 += 32) {
        union { bf16x16 v; uint4 u[2]; } a, b;
        const u16* ap = A + (size_t)(rowbase + lrow) * 64 + k0 + half * 16;
        a.u[0] = *(const uint4*)(ap);
        a.u[1] = *(const uint4*)(ap + 8);
        const u16* bp = Bt + (size_t)(colbase + lrow) * 64 + k0 + half * 16;
        b.u[0] = *(const uint4*)(bp);
        b.u[1] = *(const uint4*)(bp + 8);
        c = __builtin_amdgcn_wmma_f32_16x16x32_bf16(false, a.v, false, b.v, (short)0, c, false, false);
    }
#pragma unroll
    for (int v = 0; v < 8; ++v)
        Cout[(size_t)(rowbase + v + half * 8) * NCOL2 + colbase + lrow] = c[v];
}

// ---------------- edge scatter (layer 2): 2 edges per wave, 16 floats each ----------------
__global__ void k_scatter2(const int* __restrict__ src, const int* __restrict__ dst,
                           const int* __restrict__ et, const float* __restrict__ out2,
                           float* __restrict__ agg2) {
    int g = blockIdx.x * blockDim.x + threadIdx.x;
    int e = g >> 4, c = g & 15;
    if (e >= E_EDGES) return;
    int s = src[e], d = dst[e], r = et[e];
    float v = out2[(size_t)s * NCOL2 + r * C_DIM + c];
    atomicAdd(&agg2[(size_t)d * C_DIM + c], v);
}

__global__ void k_fin2(const float* __restrict__ agg2, const float* __restrict__ out2,
                       const float* __restrict__ deg, const float* __restrict__ bias2,
                       float* __restrict__ out) {
    int n = blockIdx.x * blockDim.x + threadIdx.x;
    if (n >= N_NODES) return;
    float di = 1.0f / fmaxf(deg[n], 1.0f);
    float v[C_DIM];
    float m = -3.4e38f;
#pragma unroll
    for (int c = 0; c < C_DIM; ++c) {
        float t = agg2[(size_t)n * C_DIM + c] * di
                + out2[(size_t)n * NCOL2 + R_REL * C_DIM + c] + bias2[c];
        v[c] = t;
        m = fmaxf(m, t);
    }
    float s = 0.0f;
#pragma unroll
    for (int c = 0; c < C_DIM; ++c) s += expf(v[c] - m);
    float lse = m + logf(s);
#pragma unroll
    for (int c = 0; c < C_DIM; ++c) out[(size_t)n * C_DIM + c] = v[c] - lse;
}

// ---------------- launcher ----------------
extern "C" void kernel_launch(void* const* d_in, const int* in_sizes, int n_in,
                              void* d_out, int out_size, void* d_ws, size_t ws_size,
                              hipStream_t stream) {
    const float* x       = (const float*)d_in[0];
    const int*   eidx    = (const int*)  d_in[1];   // [2][E]
    const int*   etype   = (const int*)  d_in[2];
    const float* basis1  = (const float*)d_in[3];
    const float* att1    = (const float*)d_in[4];
    const float* root1   = (const float*)d_in[5];
    const float* bias1   = (const float*)d_in[6];
    const float* basis2  = (const float*)d_in[7];
    const float* att2    = (const float*)d_in[8];
    const float* root2   = (const float*)d_in[9];
    const float* bias2   = (const float*)d_in[10];
    const int* src = eidx;
    const int* dst = eidx + E_EDGES;
    float* out = (float*)d_out;

    // workspace layout (bytes)
    char* base = (char*)d_ws;
    size_t off = 0;
    u16*   xb     = (u16*)  (base + off); off += (size_t)N_NODES * N_NODES * 2;   // 32 MB
    u16*   bmatT  = (u16*)  (base + off); off += (size_t)N_NODES * NCOL1 * 2;     // 4.5 MB  [576][4096]
    float* h_all  = (float*)(base + off); off += (size_t)N_NODES * NCOL1 * 4;     // 9 MB
    u16*   w2t    = (u16*)  (base + off); off += (size_t)NCOL2 * H_DIM * 2;
    u16*   h1bf   = (u16*)  (base + off); off += (size_t)N_NODES * H_DIM * 2;
    float* out2   = (float*)(base + off); off += (size_t)N_NODES * NCOL2 * 4;
    float* deg    = (float*)(base + off); off += (size_t)N_NODES * 4;             // deg/agg1/agg2
    float* agg1   = (float*)(base + off); off += (size_t)N_NODES * H_DIM * 4;     //  contiguous
    float* agg2   = (float*)(base + off); off += (size_t)N_NODES * C_DIM * 4;     //  for zeroing
    (void)ws_size; (void)in_sizes; (void)n_in; (void)out_size;

    // 1. zero accumulators (deg + agg1 + agg2, contiguous)
    {
        int nz = N_NODES * (1 + H_DIM + C_DIM);   // 331776
        k_zero<<<(nz + 255) / 256, 256, 0, stream>>>(deg, nz);
    }
    // 2. x -> bf16
    k_cvt_x<<<(N_NODES * (N_NODES / 4)) / 256, 256, 0, stream>>>((const float4*)x, (uint2*)xb);
    // 3/4. weight prep
    k_bmat1<<<(N_NODES * H_DIM) / 256, 256, 0, stream>>>(basis1, att1, root1, bmatT);
    k_bmat2<<<(NCOL2 * H_DIM + 255) / 256, 256, 0, stream>>>(basis2, att2, root2, w2t);
    // 5. degrees
    k_deg<<<E_EDGES / 256, 256, 0, stream>>>(dst, deg);
    // 6. big WMMA GEMM: h_all = x @ [W1_r... | root1]
    k_gemm1<<<dim3(N_NODES / BM, NCOL1 / BN), 256, 0, stream>>>(xb, bmatT, h_all);
    // 7. scatter-add messages (layer 1)
    k_scatter1<<<(E_EDGES * 32) / 256, 256, 0, stream>>>(src, dst, etype, h_all, agg1);
    // 8. mean + root + bias + ReLU -> bf16
    k_fin1<<<(N_NODES * H_DIM) / 256, 256, 0, stream>>>(agg1, h_all, deg, bias1, h1bf);
    // 9. WMMA GEMM 2: out2 = h1 @ [W2_r... | root2]
    k_gemm2<<<dim3(N_NODES / 128, NCOL2 / 16), 256, 0, stream>>>(h1bf, w2t, out2);
    // 10. scatter-add messages (layer 2)
    k_scatter2<<<(E_EDGES * 16) / 256, 256, 0, stream>>>(src, dst, etype, out2, agg2);
    // 11. mean + root + bias + log_softmax
    k_fin2<<<N_NODES / 256, 256, 0, stream>>>(agg2, out2, deg, bias2, out);
}
